// GraphEmbed_41618233098847
// MI455X (gfx1250) — matrix operand order repaired and checked by reading, not compiled, not run on hardware
//
#include <hip/hip_runtime.h>
#include <hip/hip_bf16.h>
#include <math.h>

// ---------------------------------------------------------------------------
// GraphEmbed pipeline for MI455X (gfx1250, wave32, WMMA bf16 16x16x32)
// ---------------------------------------------------------------------------

typedef __bf16 bh;
typedef __attribute__((ext_vector_type(16))) __bf16 v16bf;
typedef __attribute__((ext_vector_type(8)))  __bf16 v8bf;
typedef __attribute__((ext_vector_type(8)))  float  v8f;

static constexpr int kNDIM   = 128;
static constexpr int kNODES  = 256 * 128;   // B * IDX = 32768
static constexpr int kE      = 262144;
static constexpr int kGRAPHS = 256;

#define WMMA_BF16(a, b, c) \
  __builtin_amdgcn_wmma_f32_16x16x32_bf16(false, (a), false, (b), (short)0, (c), false, false)

__device__ __forceinline__ float sigmoidf_(float x) { return 1.0f / (1.0f + __expf(-x)); }

// A-matrix fragment (16x32 bf16, M x K) from row-major LDS tile, contiguous k order (k = 16h + s), matched by B.
// Lane's data = one contiguous 16-element K run -> two 16B LDS vector loads.
__device__ __forceinline__ v16bf load_a_frag(const bh* A, int lda, int m0, int k0) {
  int lane = threadIdx.x & 31;
  const bh* row = A + (size_t)(m0 + (lane & 15)) * lda + (k0 + ((lane >> 4) << 4));
  v8bf lo = *(const v8bf*)(row);        // K = kb .. kb+7
  v8bf hi = *(const v8bf*)(row + 8);    // K = kb+8 .. kb+15
  v16bf a;
#pragma unroll
  for (int h = 0; h < 8; ++h) { a[h] = lo[h]; a[h + 8] = hi[h]; }
  return a;
}

// B-matrix fragment (32x16 bf16, K x N) straight from row-major W[O][K] bf16:
// B[k][n] = W[n][k]; lane needs 16 *consecutive* K values of one row ->
// a single 32-byte contiguous per-lane read (two global_load_b128).
__device__ __forceinline__ v16bf load_b_frag(const bh* W, int ldk, int k0, int n0) {
  int lane = threadIdx.x & 31;
  const bh* p = W + (size_t)(n0 + (lane & 15)) * ldk + (k0 + ((lane >> 4) << 4));
  return *(const v16bf*)p;              // 32B aligned for ldk in {128,256,384}
}

// C/D layout: lane L, VGPR v -> M = v + 8*(L>=16), N = L%16

// ---------------------------------------------------------------------------
// Weight prep: pure f32 -> bf16 convert, layout preserved [O][K]
// ---------------------------------------------------------------------------
__global__ void __launch_bounds__(256) wprep_kernel(const float* __restrict__ W,
                                                    bh* __restrict__ Wb, int n) {
  int idx = blockIdx.x * 256 + threadIdx.x;
  if (idx < n) Wb[idx] = (bh)W[idx];
}

__global__ void __launch_bounds__(256) copy_kernel(const float4* __restrict__ in,
                                                   float4* __restrict__ out, int n4) {
  int i = blockIdx.x * 256 + threadIdx.x;
  if (i < n4) out[i] = in[i];
}

// ---------------------------------------------------------------------------
// Edge message: for 64 edges/WG, a = [hf[src], hf[tgt]] @ W.T + bias,
// scatter-add into agg[tgt] with f32 global atomics (segment_sum).
// ---------------------------------------------------------------------------
__global__ void __launch_bounds__(256) msg_kernel(const float* __restrict__ hf,
                                                  const int* __restrict__ srcIdx,
                                                  const int* __restrict__ tgtIdx,
                                                  const bh* __restrict__ Wb,
                                                  const float* __restrict__ bias,
                                                  float* __restrict__ agg) {
  constexpr int ALD = 272;                    // 256 + pad (bf16)
  __shared__ bh  Alds[64 * ALD];
  __shared__ int tgt_s[64];
  int tid = threadIdx.x;
  long e0 = (long)blockIdx.x * 64;

  {   // gather: 4 threads per edge, float4 loads, convert to bf16
    int e = tid >> 2, sub = tid & 3;
    int s  = srcIdx[e0 + e];
    int tg = tgtIdx[e0 + e];
    if (sub == 0) tgt_s[e] = tg;
    const float4* rs = (const float4*)(hf + (size_t)s  * kNDIM);
    const float4* rt = (const float4*)(hf + (size_t)tg * kNDIM);
    bh* drow = &Alds[e * ALD];
#pragma unroll
    for (int i = 0; i < 16; ++i) {
      int q = sub * 16 + i;                   // 0..31 -> hj, 32..63 -> hi
      float4 v = (q < 32) ? rs[q] : rt[q - 32];
      bh* d = drow + q * 4;
      d[0] = (bh)v.x; d[1] = (bh)v.y; d[2] = (bh)v.z; d[3] = (bh)v.w;
    }
  }
  __syncthreads();

  int wv = tid >> 5, lane = tid & 31;
  int nbase = wv * 32;                        // each wave: 2 N-tiles, 4 M-tiles
  v8f c[4][2];
#pragma unroll
  for (int mt = 0; mt < 4; ++mt)
#pragma unroll
    for (int j = 0; j < 2; ++j)
#pragma unroll
      for (int v = 0; v < 8; ++v) c[mt][j][v] = 0.0f;

  for (int ks = 0; ks < 8; ++ks) {            // K = 256
    int k0 = ks * 32;
    v16bf b[2];
#pragma unroll
    for (int j = 0; j < 2; ++j) b[j] = load_b_frag(Wb, 256, k0, nbase + j * 16);
    v16bf a[4];
#pragma unroll
    for (int mt = 0; mt < 4; ++mt) a[mt] = load_a_frag(Alds, ALD, mt * 16, k0);
#pragma unroll
    for (int mt = 0; mt < 4; ++mt)
#pragma unroll
      for (int j = 0; j < 2; ++j)
        c[mt][j] = WMMA_BF16(a[mt], b[j], c[mt][j]);
  }

  int nn = lane & 15, mh = (lane >> 4) << 3;
#pragma unroll
  for (int mt = 0; mt < 4; ++mt)
#pragma unroll
    for (int j = 0; j < 2; ++j) {
      int col = nbase + j * 16 + nn;
      float bv = bias[col];
#pragma unroll
      for (int v = 0; v < 8; ++v) {
        int m = mt * 16 + mh + v;
        int t = tgt_s[m];
        atomicAdd(&agg[(size_t)t * 256 + col], c[mt][j][v] + bv);
      }
    }
}

// ---------------------------------------------------------------------------
// GRU cell over 16 nodes/WG: gi = agg@Wih.T+bih (K=256), gh = hf@Whh.T+bhh (K=128)
// ---------------------------------------------------------------------------
__global__ void __launch_bounds__(256) gru_kernel(float* __restrict__ hf,
                                                  const float* __restrict__ agg,
                                                  const bh* __restrict__ Wih,
                                                  const bh* __restrict__ Whh,
                                                  const float* __restrict__ bih,
                                                  const float* __restrict__ bhh) {
  constexpr int ALD = 272, HLD = 144;
  __shared__ bh    Aagg[16 * ALD];
  __shared__ bh    Ahf[16 * HLD];
  __shared__ float gi_s[16][384];
  __shared__ float gh_s[16][384];
  int tid = threadIdx.x;
  long n0 = (long)blockIdx.x * 16;

  {   // stage agg (16x256) and hf (16x128) tiles as bf16
    const float4* ap = (const float4*)(agg + (size_t)n0 * 256);
#pragma unroll
    for (int i = 0; i < 4; ++i) {
      int q = tid + i * 256;                  // 1024 quads
      int m = q >> 6, qq = q & 63;
      float4 v = ap[(size_t)m * 64 + qq];
      bh* d = &Aagg[m * ALD + qq * 4];
      d[0] = (bh)v.x; d[1] = (bh)v.y; d[2] = (bh)v.z; d[3] = (bh)v.w;
    }
    const float4* hp = (const float4*)(hf + (size_t)n0 * 128);
#pragma unroll
    for (int i = 0; i < 2; ++i) {
      int q = tid + i * 256;                  // 512 quads
      int m = q >> 5, qq = q & 31;
      float4 v = hp[(size_t)m * 32 + qq];
      bh* d = &Ahf[m * HLD + qq * 4];
      d[0] = (bh)v.x; d[1] = (bh)v.y; d[2] = (bh)v.z; d[3] = (bh)v.w;
    }
  }
  __syncthreads();

  int wv = tid >> 5, lane = tid & 31;
  int nbase = wv * 48;                        // 24 N-tiles / 8 waves = 3 each
  v8f ci[3], ch[3];
#pragma unroll
  for (int j = 0; j < 3; ++j)
#pragma unroll
    for (int v = 0; v < 8; ++v) { ci[j][v] = 0.0f; ch[j][v] = 0.0f; }

  for (int ks = 0; ks < 8; ++ks) {            // gi: K = 256
    int k0 = ks * 32;
    v16bf a = load_a_frag(Aagg, ALD, 0, k0);
#pragma unroll
    for (int j = 0; j < 3; ++j) {
      v16bf b = load_b_frag(Wih, 256, k0, nbase + j * 16);
      ci[j] = WMMA_BF16(a, b, ci[j]);
    }
  }
  for (int ks = 0; ks < 4; ++ks) {            // gh: K = 128
    int k0 = ks * 32;
    v16bf a = load_a_frag(Ahf, HLD, 0, k0);
#pragma unroll
    for (int j = 0; j < 3; ++j) {
      v16bf b = load_b_frag(Whh, 128, k0, nbase + j * 16);
      ch[j] = WMMA_BF16(a, b, ch[j]);
    }
  }

  int nn = lane & 15, mh = (lane >> 4) << 3;
#pragma unroll
  for (int j = 0; j < 3; ++j) {
    int col = nbase + j * 16 + nn;
    float b1 = bih[col], b2 = bhh[col];
#pragma unroll
    for (int v = 0; v < 8; ++v) {
      int m = mh + v;
      gi_s[m][col] = ci[j][v] + b1;
      gh_s[m][col] = ch[j][v] + b2;
    }
  }
  __syncthreads();

#pragma unroll
  for (int i = 0; i < 8; ++i) {               // 16 nodes x 128 dims
    int idx = tid + i * 256;
    int m = idx >> 7, jj = idx & 127;
    float r  = sigmoidf_(gi_s[m][jj] + gh_s[m][jj]);
    float z  = sigmoidf_(gi_s[m][128 + jj] + gh_s[m][128 + jj]);
    float ng = tanhf(gi_s[m][256 + jj] + r * gh_s[m][256 + jj]);
    size_t off = (size_t)(n0 + m) * 128 + jj;
    float hv = hf[off];
    hf[off] = (1.0f - z) * ng + z * hv;
  }
}

// ---------------------------------------------------------------------------
// Row L2-normalize: one wave per 128-f32 row, shuffle reduction
// ---------------------------------------------------------------------------
__global__ void __launch_bounds__(256) norm_kernel(float* __restrict__ hf) {
  int wv = threadIdx.x >> 5, lane = threadIdx.x & 31;
  long row = (long)blockIdx.x * 8 + wv;
  float4* r = (float4*)(hf + (size_t)row * 128);
  float4 v = r[lane];
  float ss = v.x * v.x + v.y * v.y + v.z * v.z + v.w * v.w;
#pragma unroll
  for (int m = 16; m > 0; m >>= 1) ss += __shfl_xor(ss, m, 32);
  float inv = 1.0f / fmaxf(sqrtf(ss), 1e-12f);
  v.x *= inv; v.y *= inv; v.z *= inv; v.w *= inv;
  r[lane] = v;
}

// ---------------------------------------------------------------------------
// Gated pooling per graph: s = sum_m (hf[m]@fW.T + fb) * sigmoid(hf[m].gW + gb)
// then L2-normalize the 128-vector.
// ---------------------------------------------------------------------------
__global__ void __launch_bounds__(256) aggr_kernel(const float* __restrict__ hf,
                                                   const bh* __restrict__ fWb,
                                                   const float* __restrict__ fb,
                                                   const float* __restrict__ gW,
                                                   const float* __restrict__ gb,
                                                   float* __restrict__ out) {
  constexpr int HLD = 144;
  __shared__ bh    Ah[128 * HLD];             // 36.9 KB
  __shared__ float g_s[128];
  __shared__ float gw_s[128];
  __shared__ float s_part[8][128];
  __shared__ float s_vec[128];
  __shared__ float red[64];
  int tid = threadIdx.x;
  long g0 = blockIdx.x;
  const float* base = hf + (size_t)g0 * 128 * 128;

  {
    const float4* hp = (const float4*)base;
#pragma unroll
    for (int i = 0; i < 16; ++i) {
      int q = tid + i * 256;                  // 4096 quads
      int m = q >> 5, qq = q & 31;
      float4 v = hp[(size_t)m * 32 + qq];
      bh* d = &Ah[m * HLD + qq * 4];
      d[0] = (bh)v.x; d[1] = (bh)v.y; d[2] = (bh)v.z; d[3] = (bh)v.w;
    }
    if (tid < 128) gw_s[tid] = gW[tid];
  }
  __syncthreads();

  if (tid < 128) {                            // per-node gate
    float acc = gb[0];
    const bh* rowp = &Ah[tid * HLD];
#pragma unroll 8
    for (int k = 0; k < 128; ++k) acc += (float)rowp[k] * gw_s[k];
    g_s[tid] = sigmoidf_(acc);
  }
  __syncthreads();

  int wv = tid >> 5, lane = tid & 31;
  v8f c[8];
#pragma unroll
  for (int j = 0; j < 8; ++j)
#pragma unroll
    for (int v = 0; v < 8; ++v) c[j][v] = 0.0f;

  for (int ks = 0; ks < 4; ++ks) {            // K = 128; wave wv owns M-tile wv
    int k0 = ks * 32;
    v16bf a = load_a_frag(Ah, HLD, wv * 16, k0);
#pragma unroll
    for (int j = 0; j < 8; ++j) {
      v16bf b = load_b_frag(fWb, 128, k0, j * 16);
      c[j] = WMMA_BF16(a, b, c[j]);
    }
  }

  int nn = lane & 15, mh = (lane >> 4) << 3;
#pragma unroll
  for (int j = 0; j < 8; ++j) {
    int col = j * 16 + nn;
    float bv = fb[col];
    float part = 0.0f;
#pragma unroll
    for (int v = 0; v < 8; ++v) {
      int m = wv * 16 + mh + v;
      part += (c[j][v] + bv) * g_s[m];
    }
    part += __shfl_xor(part, 16, 32);         // fold two half-waves (same col)
    if (lane < 16) s_part[wv][col] = part;
  }
  __syncthreads();

  if (tid < 128) {
    float s = 0.0f;
#pragma unroll
    for (int w = 0; w < 8; ++w) s += s_part[w][tid];
    s_vec[tid] = s;
  }
  __syncthreads();
  if (tid < 64) red[tid] = s_vec[tid] * s_vec[tid] + s_vec[tid + 64] * s_vec[tid + 64];
  __syncthreads();
  for (int st = 32; st > 0; st >>= 1) {
    if (tid < st) red[tid] += red[tid + st];
    __syncthreads();
  }
  if (tid < 128) {
    float inv = 1.0f / fmaxf(sqrtf(red[0]), 1e-12f);
    out[(size_t)g0 * 128 + tid] = s_vec[tid] * inv;
  }
}

// ---------------------------------------------------------------------------
// Host-side orchestration (graph-capture safe: kernels + hipMemsetAsync only)
// ---------------------------------------------------------------------------
extern "C" void kernel_launch(void* const* d_in, const int* in_sizes, int n_in,
                              void* d_out, int out_size, void* d_ws, size_t ws_size,
                              hipStream_t stream) {
  const float* h      = (const float*)d_in[0];
  const int*   ei     = (const int*)d_in[1];
  const float* msgW   = (const float*)d_in[2];
  const float* msg_b  = (const float*)d_in[3];
  const float* msgrW  = (const float*)d_in[4];
  const float* msgr_b = (const float*)d_in[5];
  const float* gWih   = (const float*)d_in[6];
  const float* gWhh   = (const float*)d_in[7];
  const float* gbih   = (const float*)d_in[8];
  const float* gbhh   = (const float*)d_in[9];
  const float* fmW    = (const float*)d_in[10];
  const float* fmb    = (const float*)d_in[11];
  const float* gmW    = (const float*)d_in[12];
  const float* gmb    = (const float*)d_in[13];
  const float* fm2W   = (const float*)d_in[14];
  const float* fm2b   = (const float*)d_in[15];
  const float* gm2W   = (const float*)d_in[16];
  const float* gm2b   = (const float*)d_in[17];

  float* hf  = (float*)d_out;                       // [N, 128] working/output
  float* hG  = hf + (size_t)kNODES * kNDIM;         // [256, 128]
  float* hG2 = hG + (size_t)kGRAPHS * kNDIM;        // [256, 128]

  char* ws = (char*)d_ws;
  size_t off = 0;
  float* agg = (float*)(ws + off); off += (size_t)kNODES * 256 * sizeof(float);  // 32 MB
  bh* msgWb  = (bh*)(ws + off);    off += (size_t)2 * 256 * 256 * sizeof(bh);
  bh* msgrWb = (bh*)(ws + off);    off += (size_t)2 * 256 * 256 * sizeof(bh);
  bh* gihB   = (bh*)(ws + off);    off += (size_t)2 * 384 * 256 * sizeof(bh);
  bh* ghhB   = (bh*)(ws + off);    off += (size_t)2 * 384 * 128 * sizeof(bh);
  bh* fmB    = (bh*)(ws + off);    off += (size_t)128 * 128 * sizeof(bh);
  bh* fm2B   = (bh*)(ws + off);    off += (size_t)128 * 128 * sizeof(bh);

  // Weight prep: straight f32->bf16 convert, layout preserved (tiny; L2-resident)
  wprep_kernel<<<512, 256, 0, stream>>>(msgW,  msgWb,  2 * 256 * 256);
  wprep_kernel<<<512, 256, 0, stream>>>(msgrW, msgrWb, 2 * 256 * 256);
  wprep_kernel<<<768, 256, 0, stream>>>(gWih,  gihB,   2 * 384 * 256);
  wprep_kernel<<<384, 256, 0, stream>>>(gWhh,  ghhB,   2 * 384 * 128);
  wprep_kernel<<<64,  256, 0, stream>>>(fmW,   fmB,    128 * 128);
  wprep_kernel<<<64,  256, 0, stream>>>(fm2W,  fm2B,   128 * 128);

  // hf = h (do not mutate inputs)
  copy_kernel<<<(kNODES * kNDIM / 4) / 256, 256, 0, stream>>>(
      (const float4*)h, (float4*)hf, kNODES * kNDIM / 4);

  for (int l = 0; l < 2; ++l) {
    hipMemsetAsync(agg, 0, (size_t)kNODES * 256 * sizeof(float), stream);
    // forward edges: src=e[0], tgt=e[1], weights msg
    msg_kernel<<<kE / 64, 256, 0, stream>>>(hf, ei, ei + kE,
                                            msgWb + l * 65536, msg_b + l * 256, agg);
    // reversed edges: src=e[1], tgt=e[0], weights msg_rev
    msg_kernel<<<kE / 64, 256, 0, stream>>>(hf, ei + kE, ei,
                                            msgrWb + l * 65536, msgr_b + l * 256, agg);
    gru_kernel<<<kNODES / 16, 256, 0, stream>>>(hf, agg, gihB + l * 98304,
                                                ghhB + l * 49152,
                                                gbih + l * 384, gbhh + l * 384);
  }

  norm_kernel<<<kNODES / 8, 256, 0, stream>>>(hf);

  aggr_kernel<<<kGRAPHS, 256, 0, stream>>>(hf, fmB,  fmb,  gmW,  gmb,  hG);
  aggr_kernel<<<kGRAPHS, 256, 0, stream>>>(hf, fm2B, fm2b, gm2W, gm2b, hG2);
}
